// SparseAttention_30717606101524
// MI455X (gfx1250) — matrix operand order repaired and checked by reading, not compiled, not run on hardware
//
#include <hip/hip_runtime.h>
#include <hip/hip_bf16.h>
#include <math.h>

// SparseAttention (log-sparse causal): each query i attends only to keys at
// offsets d in {0,3,15,63,255,1023}. B=2, L=2048, H=16, E=64.
//
// One wave32 per (b, h, 16-query tile). Scores via v_wmma_f32_16x16x32_f16
// (diagonal of Q_tile x K_block^T per offset), softmax over <=6 logits,
// output = sum_d p_d * V[i-d] with coalesced float2 VALU FMAs.

typedef __attribute__((ext_vector_type(16))) _Float16 v16h;
typedef __attribute__((ext_vector_type(8)))  float    v8f;

#define SA_B 2
#define SA_L 2048
#define SA_H 16
#define SA_E 64
#define SA_ND 6
#define SA_TILES (SA_L / 16)           // 128
#define SA_ROWSTRIDE (SA_H * SA_E)     // 1024 floats between consecutive seq rows

__device__ __constant__ int c_offs[SA_ND] = {0, 3, 15, 63, 255, 1023};

__global__ __launch_bounds__(32)
void sparse_attn_logsparse_kernel(const float* __restrict__ Q,
                                  const float* __restrict__ K,
                                  const float* __restrict__ V,
                                  float* __restrict__ O) {
    __shared__ float sS[SA_ND * 16];   // raw scores  [d][r]
    __shared__ float sP[16 * 8];       // softmax probs [r][d] (stride 8)

    const int bid = blockIdx.x;                 // (b*H + h)*TILES + t
    const int t   = bid & (SA_TILES - 1);
    const int h   = (bid >> 7) & (SA_H - 1);
    const int b   = bid >> 11;

    const int lane = threadIdx.x;               // 0..31
    const int n    = lane & 15;                 // row/col index within 16x16 tile
    const int g    = lane >> 4;                 // half-wave group

    const int i0 = t * 16;                      // first query row of tile
    const size_t planeQ = ((size_t)b * SA_L) * SA_ROWSTRIDE + (size_t)h * SA_E;

    // ---- Build A fragment (Q tile, 16x64 as two 16x32 f16 slices) ----
    // ISA 16-bit A layout: lane group 0: K = {j} j<8, {j+8} j>=8 ; group 1: +8.
    const float* qrow = Q + planeQ + (size_t)(i0 + n) * SA_ROWSTRIDE;
    v16h aQ[2];
    #pragma unroll
    for (int s = 0; s < 2; ++s) {
        #pragma unroll
        for (int j = 0; j < 16; j += 2) {
            const int k = j + ((j < 8) ? 0 : 8) + 8 * g + 32 * s;
            const float2 qv = *reinterpret_cast<const float2*>(qrow + k);
            aQ[s][j]     = (_Float16)qv.x;
            aQ[s][j + 1] = (_Float16)qv.y;
        }
    }

    // Warm L2/L0 with the V rows this tile will gather (global_prefetch_b8).
    #pragma unroll
    for (int d = 0; d < SA_ND; ++d) {
        const int j = i0 + n - c_offs[d];
        if (j >= 0)
            __builtin_prefetch(V + planeQ + (size_t)j * SA_ROWSTRIDE, 0, 3);
    }

    // ---- Per-offset scores: diag( Q_tile x K_block^T ) via WMMA ----
    #pragma unroll
    for (int d = 0; d < SA_ND; ++d) {
        const int base = i0 - c_offs[d];
        const int rowK = base + n;
        const int rowc = rowK < 0 ? 0 : rowK;    // clamp; masked in softmax
        const float* krow = K + planeQ + (size_t)rowc * SA_ROWSTRIDE;

        // ISA 16-bit B layout (32x16): b[j] = B[K = j + (j<8 ? 0 : 8) + 8g][N = lane&15],
        // with B[k][c] = K_block[c][k]  (so each lane reads its own K row).
        v16h bK[2];
        #pragma unroll
        for (int s = 0; s < 2; ++s) {
            #pragma unroll
            for (int j = 0; j < 16; j += 2) {
                const int k = j + ((j < 8) ? 0 : 8) + 8 * g + 32 * s;
                const float2 kv = *reinterpret_cast<const float2*>(krow + k);
                bK[s][j]     = (_Float16)kv.x;
                bK[s][j + 1] = (_Float16)kv.y;
            }
        }

        v8f acc = {};
        acc = __builtin_amdgcn_wmma_f32_16x16x32_f16(
                  false, aQ[0], false, bK[0], (short)0, acc, false, false);
        acc = __builtin_amdgcn_wmma_f32_16x16x32_f16(
                  false, aQ[1], false, bK[1], (short)0, acc, false, false);

        // Diagonal of C: row r<8 -> (VGPR r, lane r); r>=8 -> (VGPR r-8, lane r+16).
        int r = -1, idx = 0;
        if (lane < 8)        { r = lane;      idx = lane;      }
        else if (lane >= 24) { r = lane - 16; idx = lane - 24; }
        if (r >= 0) {
            float val = 0.0f;
            #pragma unroll
            for (int j = 0; j < 8; ++j) val = (idx == j) ? acc[j] : val;
            sS[d * 16 + r] = val;
        }
    }
    __syncthreads();

    // ---- Softmax over <=6 logits per row (lanes 0..15, one row each) ----
    if (lane < 16) {
        const int r = lane;
        const int i = i0 + r;
        const float scale = 0.125f;              // 1/sqrt(64)
        float lg[SA_ND];
        float m = -3.0e38f;
        #pragma unroll
        for (int d = 0; d < SA_ND; ++d) {
            const bool ok = (i - c_offs[d]) >= 0;
            lg[d] = ok ? scale * sS[d * 16 + r] : -3.0e38f;
            m = fmaxf(m, lg[d]);
        }
        float sum = 0.0f;
        #pragma unroll
        for (int d = 0; d < SA_ND; ++d) {
            const bool ok = (i - c_offs[d]) >= 0;
            const float e = ok ? __expf(lg[d] - m) : 0.0f;
            lg[d] = e;
            sum += e;
        }
        const float inv = 1.0f / sum;            // d=0 always valid -> sum>0
        #pragma unroll
        for (int d = 0; d < SA_ND; ++d) sP[r * 8 + d] = lg[d] * inv;
    }
    __syncthreads();

    // ---- Output: out[i] = sum_d p[r][d] * V[i-d]  (float2 per lane) ----
    const int e0 = 2 * lane;                     // 0..62, 8B aligned
    #pragma unroll 4
    for (int r = 0; r < 16; ++r) {
        const int i = i0 + r;
        float2 acc = make_float2(0.0f, 0.0f);
        #pragma unroll
        for (int d = 0; d < SA_ND; ++d) {
            const int j = i - c_offs[d];
            if (j >= 0) {
                const float  p  = sP[r * 8 + d];
                const float2 vv = *reinterpret_cast<const float2*>(
                    V + planeQ + (size_t)j * SA_ROWSTRIDE + e0);
                acc.x = fmaf(p, vv.x, acc.x);
                acc.y = fmaf(p, vv.y, acc.y);
            }
        }
        *reinterpret_cast<float2*>(O + planeQ + (size_t)i * SA_ROWSTRIDE + e0) = acc;
    }
}

extern "C" void kernel_launch(void* const* d_in, const int* in_sizes, int n_in,
                              void* d_out, int out_size, void* d_ws, size_t ws_size,
                              hipStream_t stream) {
    (void)in_sizes; (void)n_in; (void)d_ws; (void)ws_size; (void)out_size;
    const float* Q = (const float*)d_in[0];
    const float* K = (const float*)d_in[1];
    const float* V = (const float*)d_in[2];
    float* O = (float*)d_out;
    const int blocks = SA_B * SA_H * SA_TILES;   // 4096 waves, one per tile
    sparse_attn_logsparse_kernel<<<blocks, 32, 0, stream>>>(Q, K, V, O);
}
